// DwiConv3dBatch_52063593562463
// MI455X (gfx1250) — matrix operand order repaired and checked by reading, not compiled, not run on hardware
//
#include <hip/hip_runtime.h>

typedef __attribute__((ext_vector_type(16))) _Float16 v16h;
typedef __attribute__((ext_vector_type(8)))  _Float16 v8h;
typedef __attribute__((ext_vector_type(4)))  _Float16 v4h;
typedef __attribute__((ext_vector_type(8)))  float    v8f;

#define B_     8
#define CI_    144
#define CO_    32
#define IN_    32
#define OD_    30
#define TAPS_  27
#define XS_CI  (IN_*IN_*IN_)      /* 32768 : stride between c_in planes    */
#define XS_X   (IN_*IN_)          /* 1024  : stride along X                */
#define XS_Y   IN_                /* 32    : stride along Y (Z contiguous) */
#define NCHUNK 5                  /* ceil(144/32): ci padded to 160        */
#define NFRAG  (TAPS_*NCHUNK*2)   /* 270 pre-packed B fragments            */
#define LDS_S  164                /* LDS row stride (f16): 160 + conflict pad */

// ---------------------------------------------------------------------------
// Pre-pass: repack W[32,144,27] (fp32) into per-lane WMMA B-fragment layout
// (f16) in workspace.  Fragment f = tap*10 + chunk*2 + ntile; each lane owns
// 16 contiguous f16 (32B) = column N=lane&15, K rows {0..7,16..23 | 8..15,24..31}.
// ci >= 144 (chunk-4 padding) is zero-filled.
// ---------------------------------------------------------------------------
__global__ __launch_bounds__(256)
void repack_weights_f16(const float* __restrict__ W, _Float16* __restrict__ wsW)
{
    int t = blockIdx.x * blockDim.x + threadIdx.x;
    if (t >= NFRAG * 32) return;
    int l     = t & 31;
    int f     = t >> 5;
    int tap   = f / (NCHUNK * 2);
    int rem   = f % (NCHUNK * 2);
    int chunk = rem >> 1;
    int ni    = rem & 1;
    int co    = ni * 16 + (l & 15);
    int khi   = (l >> 4) << 3;              // lanes 16..31 hold K rows 8..15, 24..31
    _Float16* dst = wsW + (size_t)t * 16;
#pragma unroll
    for (int i = 0; i < 16; ++i) {
        int ci  = chunk * 32 + (i & 7) + ((i >> 3) << 4) + khi;
        float v = (ci < CI_) ? W[((size_t)co * CI_ + ci) * TAPS_ + tap] : 0.0f;
        dst[i]  = (_Float16)v;
    }
}

// ---------------------------------------------------------------------------
// Main kernel: one wave32 per (b, ox, oy).  GEMM tile M=32(z) x N=32(co),
// K = 27 taps * 160 (ci padded).  Per (kx,ky): stage the 144x32 x-slab once
// into LDS as transposed f16 xs[z][ci] (coalesced b32 loads, ds_store_b64),
// then A fragments are two ds_load_b128 each; f32 accum in WMMA.
// ---------------------------------------------------------------------------
__global__ __launch_bounds__(32)
void dwi_conv3d_wmma(const float* __restrict__ x,
                     const _Float16* __restrict__ wsW,
                     float* __restrict__ out)
{
    __shared__ __align__(16) _Float16 xs[IN_ * LDS_S];

    const int lane   = threadIdx.x;
    const int laneLo = lane & 15;
    const int laneHi = lane >> 4;

    const int wg = blockIdx.x;              // b*900 + ox*30 + oy
    const int b  = wg / (OD_ * OD_);
    const int r  = wg % (OD_ * OD_);
    const int ox = r / OD_;
    const int oy = r % OD_;

    // zero the ci padding [144,160) of this lane's z row (written once;
    // pairs with the zero-filled chunk-4 region of the repacked B)
    {
        v8h zz = {};
        *(v8h*)(&xs[lane * LDS_S + 144]) = zz;
        *(v8h*)(&xs[lane * LDS_S + 152]) = zz;
    }

    v8f acc00 = {}; v8f acc01 = {}; v8f acc10 = {}; v8f acc11 = {};

    const int xBase = b * (CI_ * XS_CI) + lane;   // lane indexes z (coalesced)

    for (int kxy = 0; kxy < 9; ++kxy) {
        const int kx = kxy / 3;
        const int ky = kxy % 3;

        // ---- stage slab: xs[z][ci] = (f16) x[b, ci, ox+kx, oy+ky, z] ----
        const float* xrow = x + xBase + (ox + kx) * XS_X + (oy + ky) * XS_Y;
#pragma unroll 4
        for (int cb = 0; cb < CI_ / 4; ++cb) {
            const int ci = cb * 4;
            float f0 = xrow[(ci + 0) * XS_CI];
            float f1 = xrow[(ci + 1) * XS_CI];
            float f2 = xrow[(ci + 2) * XS_CI];
            float f3 = xrow[(ci + 3) * XS_CI];
            v4h h;
            h[0] = (_Float16)f0; h[1] = (_Float16)f1;
            h[2] = (_Float16)f2; h[3] = (_Float16)f3;
            *(v4h*)(&xs[lane * LDS_S + ci]) = h;      // ds_store_b64
        }
        // single wave per workgroup: DS ops are in-order within the wave,
        // compiler inserts the loadcnt/dscnt waits -> no barrier needed.

        // ---- compute: 3 kz taps consume the staged slab ----
        for (int kz = 0; kz < 3; ++kz) {
            const int tap = kxy * 3 + kz;
            int z0 = laneLo + kz;                       // <= 17
            int z1 = 16 + laneLo + kz; if (z1 > 31) z1 = 31;   // dead rows clamp

            const _Float16* ar0  = xs + z0 * LDS_S + laneHi * 8;
            const _Float16* ar1  = xs + z1 * LDS_S + laneHi * 8;
            const _Float16* brow = wsW + (size_t)(tap * (NCHUNK * 2)) * 512 + lane * 16;

#pragma unroll
            for (int chunk = 0; chunk < NCHUNK; ++chunk) {
                // A fragment: K runs {0..7} and {16..23} (+laneHi*8) along ci
                // are contiguous in xs -> two ds_load_b128 per fragment.
                v8h a0lo = *(const v8h*)(ar0 + chunk * 32);
                v8h a0hi = *(const v8h*)(ar0 + chunk * 32 + 16);
                v8h a1lo = *(const v8h*)(ar1 + chunk * 32);
                v8h a1hi = *(const v8h*)(ar1 + chunk * 32 + 16);
                v16h a0 = __builtin_shufflevector(a0lo, a0hi,
                          0,1,2,3,4,5,6,7,8,9,10,11,12,13,14,15);
                v16h a1 = __builtin_shufflevector(a1lo, a1hi,
                          0,1,2,3,4,5,6,7,8,9,10,11,12,13,14,15);
                v16h b0 = *(const v16h*)(brow + (size_t)(chunk * 2 + 0) * 512);
                v16h b1 = *(const v16h*)(brow + (size_t)(chunk * 2 + 1) * 512);
                acc00 = __builtin_amdgcn_wmma_f32_16x16x32_f16(false, a0, false, b0, (short)0, acc00, false, false);
                acc01 = __builtin_amdgcn_wmma_f32_16x16x32_f16(false, a0, false, b1, (short)0, acc01, false, false);
                acc10 = __builtin_amdgcn_wmma_f32_16x16x32_f16(false, a1, false, b0, (short)0, acc10, false, false);
                acc11 = __builtin_amdgcn_wmma_f32_16x16x32_f16(false, a1, false, b1, (short)0, acc11, false, false);
            }
        }
    }

    // C/D layout: VGPR v -> M = v + laneHi*8, N = laneLo.
#pragma unroll
    for (int v = 0; v < 8; ++v) {
        int m0 = v + laneHi * 8;            // z of M-tile 0 (always < 30)
        int m1 = 16 + m0;                   // z of M-tile 1 (30,31 masked)
        size_t o00 = ((((size_t)b * CO_ + laneLo     ) * OD_ + ox) * OD_ + oy) * OD_ + m0;
        size_t o01 = ((((size_t)b * CO_ + laneLo + 16) * OD_ + ox) * OD_ + oy) * OD_ + m0;
        out[o00] = acc00[v];
        out[o01] = acc01[v];
        if (m1 < OD_) {
            out[o00 + 16] = acc10[v];
            out[o01 + 16] = acc11[v];
        }
    }
}

// ---------------------------------------------------------------------------
extern "C" void kernel_launch(void* const* d_in, const int* in_sizes, int n_in,
                              void* d_out, int out_size, void* d_ws, size_t ws_size,
                              hipStream_t stream) {
    const float* x   = (const float*)d_in[0];   // [8,12,12,32,32,32] fp32
    const float* W   = (const float*)d_in[1];   // [32,12,12,3,3,3]  fp32
    float*       out = (float*)d_out;           // [8,32,30,30,30]   fp32
    _Float16*    wsW = (_Float16*)d_ws;         // 270 KB repacked f16 weights

    repack_weights_f16<<<dim3((NFRAG * 32 + 255) / 256), dim3(256), 0, stream>>>(W, wsW);
    dwi_conv3d_wmma<<<dim3(B_ * OD_ * OD_), dim3(32), 0, stream>>>(x, wsW, out);
}
